// DHT_Layer_26010321944865
// MI455X (gfx1250) — matrix operand order repaired and checked by reading, not compiled, not run on hardware
//
#include <hip/hip_runtime.h>
#include <stdint.h>

// ---------------------------------------------------------------------------
// DHT layer pipeline for MI455X (gfx1250, wave32, WMMA):
//   conv1x1(256->128)+BN+ReLU -> Hough transform -> 3x3 conv+BN+ReLU x2
// bf16 WMMA implicit GEMM (f32 accumulate), BN folded into weights.
// CDNA5 data movement: TDM tensor_load_to_lds for the conv1 weight tile,
// GLOBAL_LOAD_ASYNC_TO_LDS_B64 for the 3x3 halo patches, 312.5KB LDS
// accumulator for the Hough transform (one-pass reads, ds_add_f32 atomics).
// ---------------------------------------------------------------------------

#define NBATCH 8
#define CIN    256
#define CMID   128
#define HH     100
#define WW     100
#define HWPIX  10000
#define AANG   100
#define RRHO   100
#define BN_EPS 1e-5f
#define IRHO_INV (1.0f / 1.42f)   // irho = int(sqrt(100^2+100^2)+1)/100 = 1.42

#define AS1 __attribute__((address_space(1)))
#define AS3 __attribute__((address_space(3)))

typedef __attribute__((ext_vector_type(16))) __bf16 v16bf;
typedef __attribute__((ext_vector_type(8)))  float  v8f;
typedef __attribute__((ext_vector_type(4))) unsigned int tdm_v4u;
typedef __attribute__((ext_vector_type(8))) int          tdm_v8i;
typedef __attribute__((ext_vector_type(4))) int          tdm_v4i;
typedef __attribute__((vector_size(8)))     int          v2i_t;   // async b64 payload

union V16 {
    v16bf    bf;
    uint4    u4[2];
    uint16_t us[16];
};

__device__ __forceinline__ uint16_t f2bf(float f) {
    uint32_t u = __float_as_uint(f);
    uint32_t r = u + 0x7FFFu + ((u >> 16) & 1u);   // round-to-nearest-even
    return (uint16_t)(r >> 16);
}
__device__ __forceinline__ float bf2f(uint16_t h) {
    return __uint_as_float(((uint32_t)h) << 16);
}

__device__ __forceinline__ void wait_async0() {
#if __has_builtin(__builtin_amdgcn_s_wait_asynccnt)
    __builtin_amdgcn_s_wait_asynccnt(0);
#else
    asm volatile("s_wait_asynccnt 0" ::: "memory");
#endif
}

// LDS byte-offset of a generic pointer to a __shared__ object.
__device__ __forceinline__ uint32_t lds_off(void* p) {
    return (uint32_t)(uintptr_t)(AS3 void*)p;
}

// Async global->LDS copy of 8 bytes (ASYNCcnt path); falls back to a plain copy.
__device__ __forceinline__ void async_copy_b64(uint16_t* lds_dst, const uint16_t* gsrc) {
#if __has_builtin(__builtin_amdgcn_global_load_async_to_lds_b64)
    __builtin_amdgcn_global_load_async_to_lds_b64(
        (AS1 v2i_t*)(uintptr_t)gsrc, (AS3 v2i_t*)lds_dst, 0, 0);
#else
    *(uint2*)lds_dst = *(const uint2*)gsrc;
#endif
}

// ---------------------------------------------------------------------------
// Weight prep: wbf[co][k] = bf16(w[co][k] * gamma*rsqrt(var+eps)),
// bias[co] = (b-mean)*scale + beta.  (OIHW is K-linear: k = ci*9+kh*3+kw)
// ---------------------------------------------------------------------------
__global__ void prep_weights(const float* __restrict__ w, const float* __restrict__ b,
                             const float* __restrict__ g, const float* __restrict__ be,
                             const float* __restrict__ m, const float* __restrict__ v,
                             int K, uint16_t* __restrict__ wbf, float* __restrict__ bias) {
    int co = blockIdx.x;
    float s = g[co] * rsqrtf(v[co] + BN_EPS);
    for (int k = threadIdx.x; k < K; k += blockDim.x)
        wbf[co * K + k] = f2bf(w[co * K + k] * s);
    if (threadIdx.x == 0)
        bias[co] = (b[co] - m[co]) * s + be[co];
}

// ---------------------------------------------------------------------------
// Per-wave 16x16 f32 tile via v_wmma_f32_16x16x32_bf16.
// A: bf16 weight row [16co x K] (per-lane K = two contiguous 8-elt runs per
//    the CDNA5 16-bit A layout).  B: LDS [pix][K] -> same two 8-elt runs/lane.
// ---------------------------------------------------------------------------
template <bool PREF>
__device__ __forceinline__ v8f wave_gemm(const uint16_t* __restrict__ wrow,
                                         const uint16_t* __restrict__ brow,
                                         int KTOT, int half) {
    v8f acc = {0.f, 0.f, 0.f, 0.f, 0.f, 0.f, 0.f, 0.f};
    for (int kb = 0; kb < KTOT; kb += 32) {
        if (PREF && kb + 32 < KTOT)
            __builtin_prefetch(wrow + kb + 32, 0, 1);   // global_prefetch_b8
        V16 a, b;
        a.u4[0] = *(const uint4*)(wrow + kb + 8 * half);
        a.u4[1] = *(const uint4*)(wrow + kb + 16 + 8 * half);
        const uint16_t* bp = brow + kb + 8 * half;
        b.u4[0] = *(const uint4*)(bp);
        b.u4[1] = *(const uint4*)(bp + 16);
        acc = __builtin_amdgcn_wmma_f32_16x16x32_bf16(
            false, a.bf, false, b.bf, (short)0, acc, false, false);
    }
    return acc;
}

// ---------------------------------------------------------------------------
// Kernel 1: 1x1 conv (M=128co, K=256ci, N=16pix) + BN + ReLU.
// The 64KB bf16 weight tile is DMA'd into LDS once per block with the TDM
// (tensor_load_to_lds + s_wait_tensorcnt); A-operands then read from LDS.
// grid = (625, 8), block = 256 (8 waves). Dynamic LDS: 64KB W + 8KB B.
// ---------------------------------------------------------------------------
__global__ void __launch_bounds__(256)
conv1x1_bn_relu(const float* __restrict__ x, const uint16_t* __restrict__ wbf,
                const float* __restrict__ bias, uint16_t* __restrict__ out) {
    extern __shared__ uint16_t dynsm[];
    uint16_t* Wsm = dynsm;                 // [128co][256ci] bf16 = 64 KB
    uint16_t* Bsm = dynsm + CMID * CIN;    // [16pix][256ci] bf16 = 8 KB
    int n  = blockIdx.y;
    int p0 = blockIdx.x * 16;

#if __has_builtin(__builtin_amdgcn_tensor_load_to_lds)
    if (threadIdx.x < 32) {
        // D# per CDNA5 ISA: 2D tensor 128x256 of 2-byte
        // elems, tile == tensor, contiguous (stride0=256, stride1=32768).
        uint64_t ga = (uint64_t)(uintptr_t)wbf;
        tdm_v4u g0;
        g0[0] = 1u;                                   // count=1
        g0[1] = lds_off(Wsm);                         // lds_addr
        g0[2] = (uint32_t)ga;                         // global_addr[31:0]
        g0[3] = (uint32_t)((ga >> 32) & 0x01FFFFFFu) | (2u << 30);  // addr[56:32]|type=2
        tdm_v8i g1;
        g1[0] = (int)(1u << 16);                      // data_size=1 (2 bytes)
        g1[1] = (int)(256u << 16);                    // tensor_dim0 = 256
        g1[2] = (int)(128u << 16);                    // tensor_dim1 = 128
        g1[3] = (int)(256u << 16);                    // tile_dim0 = 256
        g1[4] = 128;                                  // tile_dim1 = 128
        g1[5] = 256;                                  // tensor_dim0_stride = 256
        g1[6] = (int)(32768u << 16);                  // tensor_dim1_stride = 32768
        g1[7] = 0;
        tdm_v4i g2 = {0, 0, 0, 0}, g3 = {0, 0, 0, 0};
#if __clang_major__ >= 23
        tdm_v8i g4 = {0, 0, 0, 0, 0, 0, 0, 0};
        __builtin_amdgcn_tensor_load_to_lds(g0, g1, g2, g3, g4, 0);
#else
        __builtin_amdgcn_tensor_load_to_lds(g0, g1, g2, g3, 0);
#endif
        __builtin_amdgcn_s_wait_tensorcnt(0);
    }
#else
    for (int i = threadIdx.x; i < (CMID * CIN) / 8; i += 256)
        ((uint4*)Wsm)[i] = ((const uint4*)wbf)[i];
#endif

    for (int i = threadIdx.x; i < 16 * CIN; i += 256) {
        int pix = i >> 8, k = i & 255;
        Bsm[pix * CIN + k] = f2bf(x[(n * CIN + k) * HWPIX + p0 + pix]);
    }
    __syncthreads();

    int wave = threadIdx.x >> 5, lane = threadIdx.x & 31;
    int half = lane >> 4, l = lane & 15;
    int co0 = wave * 16;

    v8f acc = wave_gemm<false>(Wsm + (co0 + l) * CIN, Bsm + l * CIN, CIN, half);

    int pix = p0 + l;
#pragma unroll
    for (int j = 0; j < 8; ++j) {
        int co = co0 + j + 8 * half;           // C/D layout: VGPR j -> M=j (+8 upper half)
        float vo = acc[j] + bias[co];
        vo = vo > 0.f ? vo : 0.f;
        out[(n * CMID + co) * HWPIX + pix] = f2bf(vo);
    }
}

// ---------------------------------------------------------------------------
// Kernel 2: Hough aggregation. Block = (n, 8-channel chunk); 8*100*100 f32
// bins in 312.5 KB LDS (only possible with CDNA5's 320KB/WGP); every input
// value read once; rho recomputed from an LDS cos/sin table; ds_add_f32.
// ---------------------------------------------------------------------------
__global__ void __launch_bounds__(256)
dht_kernel(const uint16_t* __restrict__ h1, uint16_t* __restrict__ out) {
    extern __shared__ float acc[];               // [8][100][100] + tables
    float* tcos = acc + 8 * AANG * RRHO;
    float* tsin = tcos + AANG;
    int n = blockIdx.y, c0 = blockIdx.x * 8;

    for (int i = threadIdx.x; i < 8 * AANG * RRHO; i += blockDim.x) acc[i] = 0.f;
    for (int a = threadIdx.x; a < AANG; a += blockDim.x) {
        float th = (float)a * (3.14159265358979323846f / (float)AANG);
        tcos[a] = __cosf(th) * IRHO_INV;
        tsin[a] = __sinf(th) * IRHO_INV;
    }
    __syncthreads();

    for (int p = threadIdx.x; p < HWPIX; p += blockDim.x) {
        float vals[8];
#pragma unroll
        for (int c = 0; c < 8; ++c)
            vals[c] = bf2f(h1[(n * CMID + c0 + c) * HWPIX + p]);
        float xx = (float)(p % WW - WW / 2);
        float yy = (float)(p / WW - HH / 2);
        for (int a = 0; a < AANG; ++a) {
            int r = __float2int_rn(xx * tcos[a] + yy * tsin[a]) + RRHO / 2;
            r = min(max(r, 0), RRHO - 1);
            int bin = a * RRHO + r;
#pragma unroll
            for (int c = 0; c < 8; ++c)
                atomicAdd(&acc[c * (AANG * RRHO) + bin], vals[c]);   // ds_add_f32
        }
    }
    __syncthreads();

    for (int i = threadIdx.x; i < 8 * AANG * RRHO; i += blockDim.x) {
        int c = i / (AANG * RRHO), ar = i % (AANG * RRHO);
        out[(n * CMID + c0 + c) * (AANG * RRHO) + ar] = f2bf(acc[i]);
    }
}

// ---------------------------------------------------------------------------
// Kernel 3/4: 3x3 conv + BN + ReLU, implicit GEMM (K = 128*9 = 1152).
// Stage 1: async-DMA the compact halo patch [128ci][3][32col] to LDS (24 KB)
//          with GLOBAL_LOAD_ASYNC_TO_LDS_B64 (addresses clamped in-bounds).
// Stage 2: LDS->LDS im2col expansion into Bsm[pix][k] (duplication stays
//          on-chip); out-of-range taps zeroed here, so clamping is safe.
// Stage 3: WMMA GEMM, 36 k-steps. grid = (7 r-tiles, 100 a, 8 n).
// ---------------------------------------------------------------------------
template <bool OUT_F32>
__global__ void __launch_bounds__(256)
conv3x3_bn_relu(const uint16_t* __restrict__ in, const uint16_t* __restrict__ wbf,
                const float* __restrict__ bias, uint16_t* __restrict__ outbf,
                float* __restrict__ outf) {
    const int KTOT = CMID * 9;
    int n = blockIdx.z, a = blockIdx.y, r0 = blockIdx.x * 16;
    __shared__ uint16_t patch[CMID * 3 * 32];    // [ci][kh][col], col0 <-> r0-8; 24 KB
    __shared__ uint16_t Bsm[16 * CMID * 9];      // [pix][k]; 36 KB

    // Stage 1: 384 rows x 64 bytes, 8-byte async chunks (3072 total).
    for (int idx = threadIdx.x; idx < CMID * 3 * 8; idx += 256) {
        int row = idx >> 3, ch = idx & 7;
        int ci = row / 3, kh = row - ci * 3;
        int aa = a + kh - 1;
        aa = min(max(aa, 0), AANG - 1);                 // clamp; masked in stage 2
        int c = r0 - 8 + ch * 4;
        c = min(max(c, 0), RRHO - 4);                   // keeps 8B align + in-bounds
        const uint16_t* gs = in + ((size_t)(n * CMID + ci) * AANG + aa) * RRHO + c;
        async_copy_b64(patch + ci * 96 + kh * 32 + ch * 4, gs);
    }
    wait_async0();
    __syncthreads();

    // Stage 2: im2col expansion, entirely in LDS.
    for (int kk = threadIdx.x; kk < KTOT; kk += 256) {
        int ci = kk / 9, t = kk - ci * 9;
        int kh = t / 3, kw = t - kh * 3;
        int aa = a + kh - 1;
        bool rowok = (aa >= 0) && (aa < AANG);
        const uint16_t* prow = patch + ci * 96 + kh * 32;
#pragma unroll
        for (int pix = 0; pix < 16; ++pix) {
            int rr = r0 + pix + kw - 1;
            uint16_t v = (rowok && rr >= 0 && rr < RRHO) ? prow[pix + kw + 7]
                                                         : (uint16_t)0;
            Bsm[pix * KTOT + kk] = v;
        }
    }
    __syncthreads();

    int wave = threadIdx.x >> 5, lane = threadIdx.x & 31;
    int half = lane >> 4, l = lane & 15;
    int co0 = wave * 16;

    v8f acc = wave_gemm<true>(wbf + (co0 + l) * KTOT, Bsm + l * KTOT, KTOT, half);

    int r = r0 + l;
    if (r < RRHO) {
#pragma unroll
        for (int j = 0; j < 8; ++j) {
            int co = co0 + j + 8 * half;
            float vo = acc[j] + bias[co];
            vo = vo > 0.f ? vo : 0.f;
            size_t o = ((size_t)(n * CMID + co) * AANG + a) * RRHO + r;
            if (OUT_F32) outf[o] = vo;
            else         outbf[o] = f2bf(vo);
        }
    }
}

// ---------------------------------------------------------------------------
extern "C" void kernel_launch(void* const* d_in, const int* in_sizes, int n_in,
                              void* d_out, int out_size, void* d_ws, size_t ws_size,
                              hipStream_t stream) {
    const float* x   = (const float*)d_in[0];
    const float* w1  = (const float*)d_in[1];
    const float* b1  = (const float*)d_in[2];
    const float* g1  = (const float*)d_in[3];
    const float* be1 = (const float*)d_in[4];
    const float* m1  = (const float*)d_in[5];
    const float* v1  = (const float*)d_in[6];
    const float* w2  = (const float*)d_in[7];
    const float* b2  = (const float*)d_in[8];
    const float* g2  = (const float*)d_in[9];
    const float* be2 = (const float*)d_in[10];
    const float* m2  = (const float*)d_in[11];
    const float* v2  = (const float*)d_in[12];
    const float* w3  = (const float*)d_in[13];
    const float* b3  = (const float*)d_in[14];
    const float* g3  = (const float*)d_in[15];
    const float* be3 = (const float*)d_in[16];
    const float* m3  = (const float*)d_in[17];
    const float* v3  = (const float*)d_in[18];

    uint8_t* ws = (uint8_t*)d_ws;
    uint16_t* w1bf  = (uint16_t*)(ws + 0);                 //  64 KB
    uint16_t* w2bf  = (uint16_t*)(ws + 65536);             // 288 KB
    uint16_t* w3bf  = (uint16_t*)(ws + 360448);            // 288 KB
    float*    bias1 = (float*)   (ws + 655360);
    float*    bias2 = (float*)   (ws + 655872);
    float*    bias3 = (float*)   (ws + 656384);
    uint16_t* h1    = (uint16_t*)(ws + 656896);            // 20.48 MB bf16 each
    uint16_t* hd    = (uint16_t*)(ws + 656896 + 20480000);
    uint16_t* h2    = (uint16_t*)(ws + 656896 + 2 * 20480000);

    prep_weights<<<CMID, 256, 0, stream>>>(w1, b1, g1, be1, m1, v1, CIN,      w1bf, bias1);
    prep_weights<<<CMID, 256, 0, stream>>>(w2, b2, g2, be2, m2, v2, CMID * 9, w2bf, bias2);
    prep_weights<<<CMID, 256, 0, stream>>>(w3, b3, g3, be3, m3, v3, CMID * 9, w3bf, bias3);

    size_t c1_lds = (size_t)(CMID * CIN + 16 * CIN) * sizeof(uint16_t);  // 72 KB
    conv1x1_bn_relu<<<dim3(HWPIX / 16, NBATCH), 256, c1_lds, stream>>>(x, w1bf, bias1, h1);

    size_t dht_lds = (size_t)(8 * AANG * RRHO + 2 * AANG) * sizeof(float);  // 313.3 KB
    dht_kernel<<<dim3(CMID / 8, NBATCH), 256, dht_lds, stream>>>(h1, hd);

    conv3x3_bn_relu<false><<<dim3(7, AANG, NBATCH), 256, 0, stream>>>(hd, w2bf, bias2, h2, nullptr);
    conv3x3_bn_relu<true ><<<dim3(7, AANG, NBATCH), 256, 0, stream>>>(h2, w3bf, bias3, nullptr, (float*)d_out);
}
